// LSTM_34316788695771
// MI455X (gfx1250) — matrix operand order repaired and checked by reading, not compiled, not run on hardware
//
#include <hip/hip_runtime.h>
#include <hip/hip_bf16.h>

typedef __bf16 bf16_t;
typedef __attribute__((ext_vector_type(16))) __bf16 v16bf;
typedef __attribute__((ext_vector_type(4)))  __bf16 v4bf;
typedef __attribute__((ext_vector_type(8)))  float  v8f;
typedef __attribute__((ext_vector_type(4)))  unsigned int v4u;

#define B_   256
#define T_   512
#define F_   256
#define H_   1024
#define G4   4096   // 4*H
#define KTOT 1280   // F + H
#define OUT_ 64

#define BMF  64     // batch rows per WG
#define SJ   64     // hidden units per WG (x4 gates)
#define KT   32
#define LDT  40     // padded LDS row stride in bf16 units (80 bytes)

// ---- CDNA5 async global->LDS copy (ASYNCcnt-tracked, bypasses VGPRs) ----
__device__ __forceinline__ unsigned lds_addr_of(const void* p) {
    return (unsigned)(unsigned long long)(__attribute__((address_space(3))) const void*)p;
}
__device__ __forceinline__ void async_g2l_b128(unsigned lds, unsigned long long gaddr) {
    asm volatile("global_load_async_to_lds_b128 %0, %1, off"
                 :: "v"(lds), "v"(gaddr) : "memory");
}
__device__ __forceinline__ void wait_async0() {
    asm volatile("s_wait_asynccnt 0x0" ::: "memory");
}

// -------- mask[b,t] = any(x[b,t,:] != 0) --------
__global__ void lstm_mask_kernel(const float* __restrict__ x,
                                 unsigned char* __restrict__ mask) {
    int tid = blockIdx.x * blockDim.x + threadIdx.x;   // b*T + t
    if (tid >= B_ * T_) return;
    const float4* p = reinterpret_cast<const float4*>(x + (size_t)tid * F_);
    int any = 0;
    for (int i = 0; i < F_ / 4; ++i) {
        float4 v = p[i];
        any |= (v.x != 0.f) | (v.y != 0.f) | (v.z != 0.f) | (v.w != 0.f);
    }
    mask[tid] = (unsigned char)any;
}

// -------- Bt[n][k] = bf16( k<F ? W[k][n] : U[k-F][n] )  (transposed, [4096][1280]) --------
__global__ void lstm_convB_kernel(const float* __restrict__ W,
                                  const float* __restrict__ U,
                                  bf16_t* __restrict__ Bt) {
    int idx = blockIdx.x * blockDim.x + threadIdx.x;
    if (idx >= G4 * KTOT) return;
    int n = idx / KTOT;
    int k = idx - n * KTOT;
    float v = (k < F_) ? W[(size_t)k * G4 + n] : U[(size_t)(k - F_) * G4 + n];
    Bt[idx] = (bf16_t)v;
}

// -------- fused per-step kernel: z = [x_t | h] @ [W;U], gates, masked c/h update --------
// WG owns hidden strip j0..j0+63 for ALL 4 gates and batch rows m0..m0+63,
// so the full gate nonlinearity + state update happens in-register.
__global__ __launch_bounds__(256)
void lstm_step_fused_kernel(const float* __restrict__ x,
                            const bf16_t* __restrict__ Bt,
                            const float* __restrict__ bias,
                            const unsigned char* __restrict__ mask,
                            float* __restrict__ h, float* __restrict__ c,
                            bf16_t* __restrict__ hbf, int t) {
    __shared__ __align__(16) bf16_t At[BMF * LDT];        // A tile [64 M][32 K]
    __shared__ __align__(16) bf16_t Bts[4 * SJ * LDT];    // B tiles [4 gates][64 J][32 K]

    const int tid  = threadIdx.x;
    const int lane = tid & 31;
    const int wave = tid >> 5;
    const int wm   = wave & 1;      // 2 waves along M (32 rows each)
    const int wj   = wave >> 1;     // 4 waves along J (16 cols each)
    const int j0   = blockIdx.x * SJ;
    const int m0   = blockIdx.y * BMF;
    const int lrow = lane & 15;
    const int ksel = lane >> 4;     // lane half (K split per ISA 16-bit layouts)

    v8f zero = {0.f, 0.f, 0.f, 0.f, 0.f, 0.f, 0.f, 0.f};
    v8f acc[2][4];                  // [M frag][gate]
    #pragma unroll
    for (int fm = 0; fm < 2; ++fm)
        #pragma unroll
        for (int g = 0; g < 4; ++g) acc[fm][g] = zero;

    // B-tile async staging: 4 b128 slots per thread per K-step
    auto stage_B = [&](int kb) {
        #pragma unroll
        for (int it = 0; it < 4; ++it) {
            int slot = tid + it * 256;          // 1024 slots = 256 rows x 4
            int rr   = slot >> 2;               // LDS row: g*64 + jj
            int kv   = (slot & 3) << 3;         // 8 bf16 per slot
            int ng   = (rr >> 6) * H_ + j0 + (rr & 63);   // global column
            async_g2l_b128(lds_addr_of(&Bts[rr * LDT + kv]),
                           (unsigned long long)(Bt + (size_t)ng * KTOT + kb + kv));
        }
    };

    // one K-step of WMMAs from LDS fragments
    auto mma_step = [&]() {
        union Frag { v4u u[2]; v16bf v; };
        Frag a[2], b[4];
        #pragma unroll
        for (int fm = 0; fm < 2; ++fm) {
            // A 16x32: lanes 0-15 -> K {0..7,16..23}; lanes 16-31 -> K {8..15,24..31}
            const bf16_t* pa = &At[(wm * 32 + fm * 16 + lrow) * LDT + ksel * 8];
            a[fm].u[0] = *reinterpret_cast<const v4u*>(pa);
            a[fm].u[1] = *reinterpret_cast<const v4u*>(pa + 16);
        }
        #pragma unroll
        for (int g = 0; g < 4; ++g) {
            // B 32x16: lanes 0-15 -> K {0..7,16..23}; lanes 16-31 -> K {8..15,24..31}
            const bf16_t* pb = &Bts[(g * SJ + wj * 16 + lrow) * LDT + ksel * 8];
            b[g].u[0] = *reinterpret_cast<const v4u*>(pb);
            b[g].u[1] = *reinterpret_cast<const v4u*>(pb + 16);
        }
        #pragma unroll
        for (int fm = 0; fm < 2; ++fm)
            #pragma unroll
            for (int g = 0; g < 4; ++g)
                acc[fm][g] = __builtin_amdgcn_wmma_f32_16x16x32_bf16(
                    false, a[fm].v, false, b[g].v,
                    (short)0, acc[fm][g], false, false);
    };

    // ---- phase 1: K over x_t (f32 -> bf16 convert through VGPRs) ----
    for (int kb = 0; kb < F_; kb += KT) {
        __syncthreads();
        #pragma unroll
        for (int it = 0; it < 2; ++it) {        // 512 v4 slots
            int slot = tid + it * 256;
            int row  = slot >> 3;
            int kv   = (slot & 7) << 2;
            float4 s = *reinterpret_cast<const float4*>(
                x + ((size_t)(m0 + row) * T_ + t) * F_ + kb + kv);
            v4bf d;
            d[0] = (bf16_t)s.x; d[1] = (bf16_t)s.y;
            d[2] = (bf16_t)s.z; d[3] = (bf16_t)s.w;
            *reinterpret_cast<v4bf*>(&At[row * LDT + kv]) = d;
        }
        stage_B(kb);
        wait_async0();
        __syncthreads();
        mma_step();
    }

    // ---- phase 2: K over h (pure bf16 copies, fully async to LDS) ----
    for (int kb = F_; kb < KTOT; kb += KT) {
        __syncthreads();
        {                                        // 256 b128 slots, 1 per thread
            int row = tid >> 2;
            int kv  = (tid & 3) << 3;
            async_g2l_b128(lds_addr_of(&At[row * LDT + kv]),
                           (unsigned long long)(hbf + (size_t)(m0 + row) * H_ + (kb - F_) + kv));
        }
        stage_B(kb);
        wait_async0();
        __syncthreads();
        mma_step();
    }

    // ---- epilogue: gates + masked state update, all in-register ----
    const int j = j0 + wj * 16 + lrow;          // hidden index owned by this lane
    const float bi = bias[j];
    const float bfv = bias[H_ + j];
    const float bg = bias[2 * H_ + j];
    const float bo = bias[3 * H_ + j];
    #pragma unroll
    for (int fm = 0; fm < 2; ++fm) {
        int mg = m0 + wm * 32 + fm * 16 + ksel * 8;
        #pragma unroll
        for (int r = 0; r < 8; ++r) {
            int m = mg + r;
            float zi = acc[fm][0][r] + bi;
            float zf = acc[fm][1][r] + bfv;
            float zg = acc[fm][2][r] + bg;
            float zo = acc[fm][3][r] + bo;
            float ig = 1.f / (1.f + __expf(-zi));
            float fg = 1.f / (1.f + __expf(-zf));
            float gg = tanhf(zg);
            float og = 1.f / (1.f + __expf(-zo));
            size_t idx = (size_t)m * H_ + j;
            float cv = c[idx], hv = h[idx];
            float cn = fg * cv + ig * gg;
            float hn = og * tanhf(cn);
            if (mask[m * T_ + t]) { cv = cn; hv = hn; }
            c[idx] = cv;
            h[idx] = hv;
            hbf[idx] = (bf16_t)hv;
        }
    }
}

// -------- head: y1 = leaky_relu([h,h,c] @ W1 + b1) --------
__global__ __launch_bounds__(256)
void lstm_head1_kernel(const float* __restrict__ h, const float* __restrict__ c,
                       const float* __restrict__ W1, const float* __restrict__ b1,
                       float* __restrict__ y1) {
    int n  = blockIdx.x * 256 + threadIdx.x;  // 0..1023
    int m0 = blockIdx.y * 16;
    float acc[16];
    #pragma unroll
    for (int mm = 0; mm < 16; ++mm) acc[mm] = 0.f;
    for (int k = 0; k < H_; ++k) {
        float wa = W1[(size_t)k * H_ + n] + W1[(size_t)(k + H_) * H_ + n];
        float wc = W1[(size_t)(k + 2 * H_) * H_ + n];
        #pragma unroll
        for (int mm = 0; mm < 16; ++mm)
            acc[mm] += h[(size_t)(m0 + mm) * H_ + k] * wa
                     + c[(size_t)(m0 + mm) * H_ + k] * wc;
    }
    float bb = b1[n];
    #pragma unroll
    for (int mm = 0; mm < 16; ++mm) {
        float v = acc[mm] + bb;
        y1[(size_t)(m0 + mm) * H_ + n] = (v > 0.f) ? v : 0.2f * v;
    }
}

// -------- head: out = y1 @ W2 + b2 --------
__global__ void lstm_head2_kernel(const float* __restrict__ y1,
                                  const float* __restrict__ W2,
                                  const float* __restrict__ b2,
                                  float* __restrict__ out) {
    int m = blockIdx.x;
    int o = threadIdx.x;
    float acc = 0.f;
    for (int k = 0; k < H_; ++k)
        acc += y1[(size_t)m * H_ + k] * W2[(size_t)k * OUT_ + o];
    out[m * OUT_ + o] = acc + b2[o];
}

extern "C" void kernel_launch(void* const* d_in, const int* in_sizes, int n_in,
                              void* d_out, int out_size, void* d_ws, size_t ws_size,
                              hipStream_t stream) {
    const float* x   = (const float*)d_in[0];
    const float* W   = (const float*)d_in[1];
    const float* U   = (const float*)d_in[2];
    const float* bia = (const float*)d_in[3];
    const float* W1  = (const float*)d_in[4];
    const float* b1  = (const float*)d_in[5];
    const float* W2  = (const float*)d_in[6];
    const float* b2  = (const float*)d_in[7];
    float* out = (float*)d_out;

    char* ws = (char*)d_ws;
    size_t off = 0;
    auto carve = [&](size_t bytes) -> char* {
        off = (off + 255) & ~(size_t)255;
        char* p = ws + off;
        off += bytes;
        return p;
    };
    bf16_t*        Bt   = (bf16_t*)       carve((size_t)G4 * KTOT * 2);  // 10 MB
    float*         h    = (float*)        carve((size_t)B_ * H_ * 4);    //  1 MB
    float*         c    = (float*)        carve((size_t)B_ * H_ * 4);    //  1 MB
    bf16_t*        hbf  = (bf16_t*)       carve((size_t)B_ * H_ * 2);    // .5 MB
    unsigned char* mask = (unsigned char*)carve((size_t)B_ * T_);        // 128 KB
    float*         y1   = (float*)        carve((size_t)B_ * H_ * 4);    //  1 MB
    (void)ws_size; (void)in_sizes; (void)n_in; (void)out_size;

    // zero LSTM state each call (deterministic; graph-capturable)
    (void)hipMemsetAsync(h,   0, (size_t)B_ * H_ * 4, stream);
    (void)hipMemsetAsync(c,   0, (size_t)B_ * H_ * 4, stream);
    (void)hipMemsetAsync(hbf, 0, (size_t)B_ * H_ * 2, stream);

    lstm_mask_kernel<<<(B_ * T_) / 256, 256, 0, stream>>>(x, mask);
    lstm_convB_kernel<<<(G4 * KTOT) / 256, 256, 0, stream>>>(W, U, Bt);

    dim3 sgrid(H_ / SJ, B_ / BMF);   // (16, 4)
    for (int t = 0; t < T_; ++t) {
        lstm_step_fused_kernel<<<sgrid, 256, 0, stream>>>(x, Bt, bia, mask, h, c, hbf, t);
    }

    lstm_head1_kernel<<<dim3(H_ / 256, B_ / 16), 256, 0, stream>>>(h, c, W1, b1, y1);
    lstm_head2_kernel<<<B_, OUT_, 0, stream>>>(y1, W2, b2, out);
}
